// QRotationCrossAttention_46952582480157
// MI455X (gfx1250) — matrix operand order repaired and checked by reading, not compiled, not run on hardware
//
#include <hip/hip_runtime.h>
#include <hip/hip_bf16.h>

#define DI __device__ __forceinline__

typedef __attribute__((ext_vector_type(16))) __bf16 v16bf;
typedef __attribute__((ext_vector_type(8)))  float  v8f;
typedef __attribute__((ext_vector_type(4)))  int    v4i;

#if __has_builtin(__builtin_amdgcn_global_load_async_to_lds_b128) && \
    __has_builtin(__builtin_amdgcn_s_wait_asynccnt)
#define ATHENA_ASYNC 1
#endif

namespace {

constexpr int kB = 4;       // batch
constexpr int kC = 512;     // channels
constexpr int kS = 16384;   // Q*H*W
constexpr int kG = 128;     // C/4 (head dim)
constexpr int LDT = 40;     // padded LDS row stride in halfwords (80B: multiple of 16B, staggers banks)
constexpr int kTileE = 128 * LDT;   // one LDS tile in halfwords

typedef __attribute__((address_space(1))) v4i gv4i;   // global int4
typedef __attribute__((address_space(3))) v4i lv4i;   // LDS int4

DI unsigned short f2bf(float f) {
  unsigned u = __builtin_bit_cast(unsigned, f);
  u += 0x7FFFu + ((u >> 16) & 1u);   // round-to-nearest-even
  return (unsigned short)(u >> 16);
}
DI float bf2f(unsigned short h) {
  unsigned u = ((unsigned)h) << 16;
  return __builtin_bit_cast(float, u);
}

// 16-bit A-matrix 16x32 fragment (ISA 7.12.2): lane m = L%16, half = L/16.
// VGPR0..3 hold K = 8*half..+7, VGPR4..7 hold K = 16+8*half..+7 -> two 16B LDS chunks.
DI v16bf load_fragA(const unsigned short* s, int row, int lane) {
  int m = row + (lane & 15);
  int half = lane >> 4;
  const unsigned short* p = s + m * LDT + half * 8;
  union { v16bf v; float4 f[2]; } u;
  u.f[0] = *reinterpret_cast<const float4*>(p);
  u.f[1] = *reinterpret_cast<const float4*>(p + 16);
  return u.v;
}
// 16-bit B-matrix 32x16 fragment: lane n = L%16; VGPR0..3 hold K = 8*half..+7, VGPR4..7 hold K = 16+8*half..+7.
DI v16bf load_fragB(const unsigned short* s, int col, int lane) {
  int n = col + (lane & 15);
  int half = lane >> 4;
  const unsigned short* p = s + n * LDT + half * 8;
  union { v16bf v; float4 f[2]; } u;
  u.f[0] = *reinterpret_cast<const float4*>(p);
  u.f[1] = *reinterpret_cast<const float4*>(p + 16);
  return u.v;
}

// Stage 128 rows x 32 bf16 (contiguous per row) global -> LDS tile [128][LDT].
// Async path: 512 x 16B chunks via GLOBAL_LOAD_ASYNC_TO_LDS_B128 (2 issues/wave, ASYNCcnt).
DI void stage_rows(unsigned short* sDst, const unsigned short* __restrict__ gSrc,
                   size_t rowStrideElems, int tid) {
#ifdef ATHENA_ASYNC
  for (int i = tid; i < 512; i += 256) {
    int r = i >> 2, c = (i & 3) * 8;     // 8 bf16 = 16B chunk
    gv4i* g = (gv4i*)(gSrc + (size_t)r * rowStrideElems + c);
    lv4i* l = (lv4i*)(sDst + r * LDT + c);
    __builtin_amdgcn_global_load_async_to_lds_b128(g, l, 0, 0);
  }
#else
  for (int i = tid; i < 128 * 32; i += 256) {
    int r = i >> 5, k = i & 31;
    sDst[r * LDT + k] = gSrc[(size_t)r * rowStrideElems + k];
  }
#endif
}

enum GemmMode {
  M_A1 = 0,     // a1 = relu(Wa1 * x + ba1)           -> bf16 [b][128][S]
  M_QK = 1,     // (W * x + b) * gate                  -> bf16 head-major [b][h][g][S]
  M_V = 2,      // W * x + b                           -> bf16 head-major
  M_LOGITS = 3, // split-K partials of qh * kh^T       -> f32 [split][bh][128][128]
  M_APPLY = 4,  // attnP * vh                          -> bf16 [bh][g][S]
  M_WO = 5,     // Wo * oh(perm) + bo                  -> f32 [b][c][S]
};

template <int MODE>
__global__ __launch_bounds__(256)
void gemm_wmma_kernel(const float* __restrict__ Af, const unsigned short* __restrict__ Ah,
                      const unsigned short* __restrict__ Bh,
                      const float* __restrict__ bias, const float* __restrict__ gate,
                      float* __restrict__ outF, unsigned short* __restrict__ outH,
                      int Kdim) {
  __shared__ __align__(16) unsigned short sA[2 * kTileE];   // double-buffered
  __shared__ __align__(16) unsigned short sB[2 * kTileE];

  const int tid = threadIdx.x;
  const int lane = tid & 31;
  const int wv = tid >> 5;
  const int wm = (wv & 3) * 32;   // wave row offset inside 128-row tile
  const int wn = (wv >> 2) * 64;  // wave col offset inside 128-col tile
  const int mt = blockIdx.y;
  const int nt = (MODE == M_LOGITS) ? 0 : blockIdx.x;
  const int z = blockIdx.z;       // batch b, or bh = b*4+h for attention modes

  // per-wave async issues per tile (2 per stage_rows call; in-order completion)
  constexpr int kAsyncPerTile = (MODE == M_LOGITS) ? 4 : ((MODE == M_WO) ? 0 : 2);

  const v8f z8 = {0.f, 0.f, 0.f, 0.f, 0.f, 0.f, 0.f, 0.f};
  v8f acc[2][4];
#pragma unroll
  for (int i = 0; i < 2; ++i)
#pragma unroll
    for (int j = 0; j < 4; ++j) acc[i][j] = z8;

  int k0 = 0, k1 = Kdim;
  if constexpr (MODE == M_LOGITS) {   // deterministic split-K over blockIdx.x (8 chunks)
    int chunk = Kdim >> 3;
    k0 = blockIdx.x * chunk;
    k1 = k0 + chunk;
  }

  // Stage one full k-step (A and B tiles) into LDS buffer `buf`.
  auto stage_tile = [&](int kk, int buf) {
    unsigned short* dA = sA + buf * kTileE;
    unsigned short* dB = sB + buf * kTileE;
    // ---- A tile [128 rows][32 k] ----
    if constexpr (MODE == M_LOGITS) {
      stage_rows(dA, Ah + (size_t)z * kG * kS + kk, kS, tid);          // qh rows (bf16, async)
    } else if constexpr (MODE == M_APPLY) {
      stage_rows(dA, Ah + (size_t)z * kG * kG + kk, kG, tid);          // attnP rows (bf16, async)
    } else {
      for (int i = tid; i < 128 * 8; i += 256) {                       // f32 weights -> packed bf16
        int r = i >> 3, kq = (i & 7) * 4;
        float4 w = *reinterpret_cast<const float4*>(Af + (size_t)(mt * 128 + r) * Kdim + kk + kq);
        uint2 pk;
        pk.x = (unsigned)f2bf(w.x) | ((unsigned)f2bf(w.y) << 16);
        pk.y = (unsigned)f2bf(w.z) | ((unsigned)f2bf(w.w) << 16);
        *reinterpret_cast<uint2*>(dA + r * LDT + kq) = pk;
      }
    }
    // ---- Bt tile [128 n][32 k] ----
    if constexpr (MODE == M_LOGITS) {
      stage_rows(dB, Bh + (size_t)z * kG * kS + kk, kS, tid);          // kh rows (bf16, async)
    } else if constexpr (MODE == M_A1 || MODE == M_QK || MODE == M_V) {
      // pre-transposed bf16 activations xbt[b][s][c]: Bt rows are contiguous (async)
      stage_rows(dB, Bh + (size_t)z * kS * kC + (size_t)(nt * 128) * kC + kk, kC, tid);
    } else if constexpr (MODE == M_APPLY) {
      for (int i = tid; i < 32 * 64; i += 256) {                       // vh transpose (u32 loads)
        int k = i >> 6, n = (i & 63) * 2;
        unsigned v = *reinterpret_cast<const unsigned*>(
            Bh + (size_t)z * kG * kS + (size_t)(kk + k) * kS + nt * 128 + n);
        dB[n * LDT + k] = (unsigned short)v;
        dB[(n + 1) * LDT + k] = (unsigned short)(v >> 16);
      }
    } else {  // M_WO: head-permuted channel rows, transpose (u32 loads)
      for (int i = tid; i < 32 * 64; i += 256) {
        int k = i >> 6, n = (i & 63) * 2;
        int c = kk + k;
        unsigned v = *reinterpret_cast<const unsigned*>(
            Bh + (size_t)z * kC * kS + (size_t)((c & 3) * kG + (c >> 2)) * kS + nt * 128 + n);
        dB[n * LDT + k] = (unsigned short)v;
        dB[(n + 1) * LDT + k] = (unsigned short)(v >> 16);
      }
    }
  };

  const int steps = (k1 - k0) >> 5;
  stage_tile(k0, 0);                       // prologue: tile 0 in flight
  for (int t = 0; t < steps; ++t) {
    const int cur = t & 1;
    if (t + 1 < steps) {
      stage_tile(k0 + ((t + 1) << 5), cur ^ 1);   // issue next tile into other buffer
#ifdef ATHENA_ASYNC
      __builtin_amdgcn_s_wait_asynccnt(kAsyncPerTile);  // drain to: only next tile outstanding
#endif
    } else {
#ifdef ATHENA_ASYNC
      __builtin_amdgcn_s_wait_asynccnt(0);
#endif
    }
    __syncthreads();                       // current tile visible to all waves

    const unsigned short* cA = sA + cur * kTileE;
    const unsigned short* cB = sB + cur * kTileE;
    v16bf fa0 = load_fragA(cA, wm, lane);
    v16bf fa1 = load_fragA(cA, wm + 16, lane);
    v16bf fb[4];
#pragma unroll
    for (int j = 0; j < 4; ++j) fb[j] = load_fragB(cB, wn + j * 16, lane);
#pragma unroll
    for (int j = 0; j < 4; ++j) {
      acc[0][j] = __builtin_amdgcn_wmma_f32_16x16x32_bf16(false, fa0, false, fb[j], (short)0, acc[0][j], false, false);
      acc[1][j] = __builtin_amdgcn_wmma_f32_16x16x32_bf16(false, fa1, false, fb[j], (short)0, acc[1][j], false, false);
    }
    __syncthreads();                       // all reads done before buffer is restaged
  }

  // ---- epilogue: C/D layout (ISA 7.12.2): m = r + 8*(lane/16), n = lane%16 ----
  const int half = lane >> 4, nl = lane & 15;
#pragma unroll
  for (int i = 0; i < 2; ++i)
#pragma unroll
    for (int j = 0; j < 4; ++j)
#pragma unroll
      for (int r = 0; r < 8; ++r) {
        int m = mt * 128 + wm + i * 16 + r + half * 8;
        int n = nt * 128 + wn + j * 16 + nl;
        float v = acc[i][j][r];
        if constexpr (MODE == M_A1) {
          v = fmaxf(v + bias[m], 0.0f);
          outH[(size_t)z * kG * kS + (size_t)m * kS + n] = f2bf(v);
        } else if constexpr (MODE == M_QK) {
          v = (v + bias[m]) * gate[((size_t)z * 4 + (m & 3)) * kS + n];
          outH[(size_t)z * kC * kS + (size_t)((m & 3) * kG + (m >> 2)) * kS + n] = f2bf(v);
        } else if constexpr (MODE == M_V) {
          v += bias[m];
          outH[(size_t)z * kC * kS + (size_t)((m & 3) * kG + (m >> 2)) * kS + n] = f2bf(v);
        } else if constexpr (MODE == M_LOGITS) {
          outF[(((size_t)blockIdx.x * 16 + z) * kG + m) * kG + n] = v;   // unscaled partial
        } else if constexpr (MODE == M_APPLY) {
          outH[(size_t)z * kG * kS + (size_t)m * kS + n] = f2bf(v);
        } else {  // M_WO
          outF[(size_t)z * kC * kS + (size_t)m * kS + n] = v + bias[m];
        }
      }
}

// f32 [C][S] per batch -> bf16 [S][C] per batch (tiled transpose + convert).
__global__ __launch_bounds__(256)
void transpose_cvt_kernel(const float* __restrict__ x, unsigned short* __restrict__ xt) {
  __shared__ unsigned short t[64 * 65];
  int b = blockIdx.z;
  int c0 = blockIdx.y * 64;
  int s0 = blockIdx.x * 64;
  const float* src = x + (size_t)b * kC * kS;
  unsigned short* dst = xt + (size_t)b * kS * kC;
  for (int i = threadIdx.x; i < 64 * 64; i += 256) {
    int ci = i >> 6, si = i & 63;                         // coalesced along s
    t[ci * 65 + si] = f2bf(src[(size_t)(c0 + ci) * kS + s0 + si]);
  }
  __syncthreads();
  for (int i = threadIdx.x; i < 64 * 64; i += 256) {
    int si = i >> 6, ci = i & 63;                         // coalesced along c
    dst[(size_t)(s0 + si) * kC + c0 + ci] = t[ci * 65 + si];
  }
}

// gate head: per pixel, 4 dots of length 128 over a1 (bf16), then softmax over the 4 gates.
__global__ __launch_bounds__(256)
void gate_softmax_kernel(const unsigned short* __restrict__ a1, const float* __restrict__ Wa2,
                         const float* __restrict__ ba2, float* __restrict__ awOut) {
  int idx = blockIdx.x * 256 + threadIdx.x;    // B*S = 65536 pixels
  int b = idx >> 14;
  int s = idx & (kS - 1);
  const unsigned short* base = a1 + (size_t)b * kG * kS + s;
  float a0 = ba2[0], a1c = ba2[1], a2 = ba2[2], a3 = ba2[3];
  for (int j = 0; j < kG; ++j) {
    float v = bf2f(base[(size_t)j * kS]);
    a0 = fmaf(Wa2[j], v, a0);
    a1c = fmaf(Wa2[kG + j], v, a1c);
    a2 = fmaf(Wa2[2 * kG + j], v, a2);
    a3 = fmaf(Wa2[3 * kG + j], v, a3);
  }
  float mx = fmaxf(fmaxf(a0, a1c), fmaxf(a2, a3));
  float e0 = __expf(a0 - mx), e1 = __expf(a1c - mx), e2 = __expf(a2 - mx), e3 = __expf(a3 - mx);
  float inv = 1.0f / (e0 + e1 + e2 + e3);
  size_t o = (size_t)b * 4 * kS + s;
  awOut[o] = e0 * inv;
  awOut[o + kS] = e1 * inv;
  awOut[o + 2 * (size_t)kS] = e2 * inv;
  awOut[o + 3 * (size_t)kS] = e3 * inv;
}

// sum 8 split-K partials, scale, softmax over f (row length 128); one block per (bh,g) row.
__global__ __launch_bounds__(128)
void attn_softmax_kernel(const float* __restrict__ part, unsigned short* __restrict__ P) {
  int row = blockIdx.x;            // bh*128 + g, 2048 rows
  int n = threadIdx.x;             // 0..127 = f
  int z = row >> 7, g = row & 127;
  float v = 0.f;
#pragma unroll
  for (int p = 0; p < 8; ++p)
    v += part[(((size_t)p * 16 + z) * kG + g) * kG + n];
  v *= 0.08838834764831845f;       // 1/sqrt(128)
  __shared__ float red[128];
  red[n] = v;
  __syncthreads();
  for (int t = 64; t > 0; t >>= 1) {
    if (n < t) red[n] = fmaxf(red[n], red[n + t]);
    __syncthreads();
  }
  float mx = red[0];
  __syncthreads();
  float e = __expf(v - mx);
  red[n] = e;
  __syncthreads();
  for (int t = 64; t > 0; t >>= 1) {
    if (n < t) red[n] += red[n + t];
    __syncthreads();
  }
  P[(size_t)row * kG + n] = f2bf(e / red[0]);
}

// per-(c, q) mean/var over (b, h, w): 2048 groups x 16384 elements.
__global__ __launch_bounds__(256)
void stats_kernel(const float* __restrict__ o2, float* __restrict__ meanBuf, float* __restrict__ rstdBuf) {
  int grp = blockIdx.x;            // c*4 + qi
  int c = grp >> 2, qi = grp & 3;
  float s = 0.f, s2 = 0.f;
  for (int i = threadIdx.x; i < kB * 4096; i += 256) {
    int b = i >> 12, hw = i & 4095;
    float v = o2[((size_t)b * kC + c) * kS + qi * 4096 + hw];
    s += v;
    s2 += v * v;
  }
  __shared__ float r1[256], r2[256];
  r1[threadIdx.x] = s;
  r2[threadIdx.x] = s2;
  __syncthreads();
  for (int t = 128; t > 0; t >>= 1) {
    if (threadIdx.x < t) {
      r1[threadIdx.x] += r1[threadIdx.x + t];
      r2[threadIdx.x] += r2[threadIdx.x + t];
    }
    __syncthreads();
  }
  if (threadIdx.x == 0) {
    float m = r1[0] * (1.0f / 16384.0f);
    float var = r2[0] * (1.0f / 16384.0f) - m * m;
    meanBuf[grp] = m;
    rstdBuf[grp] = rsqrtf(var + 1e-5f);
  }
}

__global__ __launch_bounds__(256)
void norm_kernel(const float* __restrict__ o2, const float* __restrict__ meanBuf,
                 const float* __restrict__ rstdBuf, const float* __restrict__ gamma,
                 const float* __restrict__ beta, float* __restrict__ out) {
  size_t idx = (size_t)blockIdx.x * 256 + threadIdx.x;   // 4*512*16384 exact
  int s = (int)(idx & (kS - 1));
  int c = (int)((idx >> 14) & (kC - 1));
  int grp = c * 4 + (s >> 12);
  out[idx] = (o2[idx] - meanBuf[grp]) * rstdBuf[grp] * gamma[c] + beta[c];
}

}  // namespace

extern "C" void kernel_launch(void* const* d_in, const int* in_sizes, int n_in,
                              void* d_out, int out_size, void* d_ws, size_t ws_size,
                              hipStream_t stream) {
  const float* x1 = (const float*)d_in[0];
  const float* x2 = (const float*)d_in[1];
  const float* Wq = (const float*)d_in[2];
  const float* bq = (const float*)d_in[3];
  const float* Wk = (const float*)d_in[4];
  const float* bk = (const float*)d_in[5];
  const float* Wv = (const float*)d_in[6];
  const float* bv = (const float*)d_in[7];
  const float* Wa1 = (const float*)d_in[8];
  const float* ba1 = (const float*)d_in[9];
  const float* Wa2 = (const float*)d_in[10];
  const float* ba2 = (const float*)d_in[11];
  const float* Wo = (const float*)d_in[12];
  const float* bo = (const float*)d_in[13];
  const float* gamma = (const float*)d_in[14];
  const float* beta = (const float*)d_in[15];
  float* out = (float*)d_out;

  // workspace layout (bytes); peak ~337 MB with reuse
  char* ws = (char*)d_ws;
  unsigned short* a1 = (unsigned short*)(ws + 0);            //  16.8 MB  bf16 [b][128][S]
  float* awBuf = (float*)(ws + 16777216);                    //   1.0 MB  f32  [b][4][S]
  unsigned short* qg = (unsigned short*)(ws + 17825792);     //  67.1 MB  bf16 head-major
  unsigned short* kg = (unsigned short*)(ws + 84934656);     //  67.1 MB
  unsigned short* vh = (unsigned short*)(ws + 152043520);    //  67.1 MB
  unsigned short* xb1t = (unsigned short*)(ws + 219152384);  //  67.1 MB  bf16 [b][s][c] (dead after q)
  unsigned short* xb2t = (unsigned short*)(ws + 286261248);  //  67.1 MB  bf16 [b][s][c] (dead after v)
  float* attnPart = (float*)(ws + 219152384);                //   8.4 MB  overlays dead xb1t
  unsigned short* attnP = (unsigned short*)(ws + 227540992); //   0.5 MB  overlays dead xb1t
  float* meanBuf = (float*)(ws + 228065280);                 //   8 KB    overlays dead xb1t
  float* rstdBuf = (float*)(ws + 228073472);                 //   8 KB    overlays dead xb1t
  unsigned short* oh = qg;                                   // reuse: qg dead after logits
  float* o2 = (float*)(ws + 84934656);                       // reuse: kg+vh dead before Wo writes

  dim3 blk(256);
  // 0) x1, x2 -> bf16 transposed panels [b][s][c]
  transpose_cvt_kernel<<<dim3(256, 8, 4), blk, 0, stream>>>(x1, xb1t);
  transpose_cvt_kernel<<<dim3(256, 8, 4), blk, 0, stream>>>(x2, xb2t);
  // 1) a1 = relu(Wa1 x1 + ba1)
  gemm_wmma_kernel<M_A1><<<dim3(128, 1, 4), blk, 0, stream>>>(Wa1, nullptr, xb1t, ba1, nullptr, nullptr, a1, 512);
  // 2) gate weights aw = softmax(Wa2 a1 + ba2)
  gate_softmax_kernel<<<dim3(256), blk, 0, stream>>>(a1, Wa2, ba2, awBuf);
  // 3) gated q, k; plain v (all stored bf16 head-major)
  gemm_wmma_kernel<M_QK><<<dim3(128, 4, 4), blk, 0, stream>>>(Wq, nullptr, xb1t, bq, awBuf, nullptr, qg, 512);
  gemm_wmma_kernel<M_QK><<<dim3(128, 4, 4), blk, 0, stream>>>(Wk, nullptr, xb2t, bk, awBuf, nullptr, kg, 512);
  gemm_wmma_kernel<M_V><<<dim3(128, 4, 4), blk, 0, stream>>>(Wv, nullptr, xb2t, bv, nullptr, nullptr, vh, 512);
  // 4) attention logits (split-K=8, deterministic partials)
  gemm_wmma_kernel<M_LOGITS><<<dim3(8, 1, 16), blk, 0, stream>>>(nullptr, qg, kg, nullptr, nullptr, attnPart, nullptr, 16384);
  // 5) reduce partials + scale + softmax over f
  attn_softmax_kernel<<<dim3(2048), dim3(128), 0, stream>>>(attnPart, attnP);
  // 6) out_h = attn * vh  (writes into qg region)
  gemm_wmma_kernel<M_APPLY><<<dim3(128, 1, 16), blk, 0, stream>>>(nullptr, attnP, vh, nullptr, nullptr, nullptr, oh, 128);
  // 7) o2 = Wo * oh(perm) + bo  (f32, overlays kg+vh)
  gemm_wmma_kernel<M_WO><<<dim3(128, 4, 4), blk, 0, stream>>>(Wo, nullptr, oh, bo, nullptr, o2, nullptr, 512);
  // 8) IQBN stats + normalize
  stats_kernel<<<dim3(2048), blk, 0, stream>>>(o2, meanBuf, rstdBuf);
  norm_kernel<<<dim3(131072), blk, 0, stream>>>(o2, meanBuf, rstdBuf, gamma, beta, out);
}
